// FrameTransformerDecoder_8581344658024
// MI455X (gfx1250) — matrix operand order repaired and checked by reading, not compiled, not run on hardware
//
#include <hip/hip_runtime.h>

// ---------------------------------------------------------------------------
// FrameTransformerDecoder for MI455X (gfx1250, wave32, WMMA).
// All heavy matmuls: v_wmma_f32_16x16x32_f16 (f32 accumulate), 128x128x32
// block tiles, float4 staging, register double-buffering.
// GEMM is templated on <BT, EPI> so the hot loop has no runtime branches.
// ---------------------------------------------------------------------------

typedef __attribute__((ext_vector_type(16))) _Float16 v16h;
typedef __attribute__((ext_vector_type(8)))  _Float16 v8h;
typedef __attribute__((ext_vector_type(4)))  _Float16 v4h;
typedef __attribute__((ext_vector_type(8)))  float    v8f;

#define TBM 128
#define TBN 128
#define TBK 32
#define LDSPAD 8   // f16 pad -> row stride 40 halves = 80B (8B/16B aligned chunks)

struct GemmP {
  const float* A;   // (M,K) row-major per z-slice
  const float* B;   // BT=1: (N,K) row-major; BT=0: (K,N) row-major
  float*       C;   // (M,N) row-major per z-slice
  int M, N, K;
  int lda, ldb, ldc;
  int zdiv;                       // z decomposed as (z/zdiv, z%zdiv)
  long long sA1, sA2;             // offsets in elements
  long long sB1, sB2;
  long long sC1, sC2;
  int bTrans;
  int epi;                        // 0=none 1=relu 2=silu
  float alpha;
};

template <int BT, int EPI>
__global__ __launch_bounds__(256) void gemm_wmma_kernel(GemmP p) {
  __shared__ _Float16 As[TBM][TBK + LDSPAD];
  __shared__ _Float16 Bs[TBN][TBK + LDSPAD];

  const int z = blockIdx.z;
  const long long zq = z / p.zdiv, zr = z % p.zdiv;
  const float* A = p.A + zq * p.sA1 + zr * p.sA2;
  const float* B = p.B + zq * p.sB1 + zr * p.sB2;
  float*       C = p.C + zq * p.sC1 + zr * p.sC2;

  const int m0 = blockIdx.y * TBM;
  const int n0 = blockIdx.x * TBN;
  const int tid  = threadIdx.x;
  const int lane = tid & 31;
  const int wid  = tid >> 5;     // 8 waves
  const int wm   = wid >> 1;     // 0..3  (32-row strip)
  const int wn   = wid & 1;      // 0..1  (64-col strip)

  const int lrow  = lane & 15;
  const int khalf = (lane >> 4) << 3;   // A: 0 or 8
  const int kb16  = (lane >> 4) << 4;   // 0 or 16; unused, B reads use khalf too

  // ---- staging geometry (hoisted out of the k-loop) ----
  const int c4 = tid & 7;        // float4 chunk in k: cols c4*4 .. c4*4+3
  const int r0 = tid >> 3;       // row 0..31 (+32 per chunk index)
  const int n4  = tid & 31;      // BT=0: n chunk, cols n4*4..+3
  const int kk0 = tid >> 5;      // BT=0: k row 0..7 (+8 per chunk index)

  const float* aPtr = A + (long long)(m0 + r0) * p.lda + (c4 << 2);
  const float* bPtr = BT ? (B + (long long)(n0 + r0) * p.ldb + (c4 << 2))
                         : (B + (long long)kk0 * p.ldb + n0 + (n4 << 2));
  const long long aStep = 32ll * p.lda;                 // per-chunk row stride
  const long long bStep = BT ? 32ll * p.ldb : 8ll * p.ldb;
  const long long bAdv  = BT ? (long long)TBK : 32ll * p.ldb;  // per k-tile advance

  v8f acc[2][4] = {};

  float4 aR[4], bR[4];
#pragma unroll
  for (int i = 0; i < 4; ++i) aR[i] = *(const float4*)(aPtr + i * aStep);
#pragma unroll
  for (int i = 0; i < 4; ++i) bR[i] = *(const float4*)(bPtr + i * bStep);
  aPtr += TBK; bPtr += bAdv;

  for (int k0 = 0; k0 < p.K; k0 += TBK) {
    // ---- store staged registers to LDS (packed f16) ----
#pragma unroll
    for (int i = 0; i < 4; ++i) {
      v4h h;
      h[0] = (_Float16)aR[i].x; h[1] = (_Float16)aR[i].y;
      h[2] = (_Float16)aR[i].z; h[3] = (_Float16)aR[i].w;
      *(v4h*)&As[r0 + i * 32][c4 << 2] = h;
    }
    if (BT) {
#pragma unroll
      for (int i = 0; i < 4; ++i) {
        v4h h;
        h[0] = (_Float16)bR[i].x; h[1] = (_Float16)bR[i].y;
        h[2] = (_Float16)bR[i].z; h[3] = (_Float16)bR[i].w;
        *(v4h*)&Bs[r0 + i * 32][c4 << 2] = h;
      }
    } else {
#pragma unroll
      for (int i = 0; i < 4; ++i) {
        int kk = kk0 + i * 8;
        Bs[(n4 << 2) + 0][kk] = (_Float16)bR[i].x;
        Bs[(n4 << 2) + 1][kk] = (_Float16)bR[i].y;
        Bs[(n4 << 2) + 2][kk] = (_Float16)bR[i].z;
        Bs[(n4 << 2) + 3][kk] = (_Float16)bR[i].w;
      }
    }
    __syncthreads();

    // ---- kick off next tile's global loads (overlap with WMMA) ----
    if (k0 + TBK < p.K) {
#pragma unroll
      for (int i = 0; i < 4; ++i) aR[i] = *(const float4*)(aPtr + i * aStep);
#pragma unroll
      for (int i = 0; i < 4; ++i) bR[i] = *(const float4*)(bPtr + i * bStep);
      aPtr += TBK; bPtr += bAdv;
    }

    // ---- fragments per ISA VGPR layouts + 2x4 WMMA ----
    v16h afr[2], bfr[4];
#pragma unroll
    for (int mi = 0; mi < 2; ++mi) {
      const _Float16* ap = &As[wm * 32 + mi * 16 + lrow][0];
      v8h lo = *(const v8h*)(ap + khalf);          // K khalf..khalf+7
      v8h hi = *(const v8h*)(ap + 16 + khalf);     // K 16+khalf..
      afr[mi] = __builtin_shufflevector(lo, hi, 0,1,2,3,4,5,6,7,8,9,10,11,12,13,14,15);
    }
#pragma unroll
    for (int ni = 0; ni < 4; ++ni) {
      const _Float16* bp = &Bs[wn * 64 + ni * 16 + lrow][0];
      v8h lo = *(const v8h*)(bp + khalf);          // K khalf..khalf+7
      v8h hi = *(const v8h*)(bp + 16 + khalf);     // K 16+khalf..
      bfr[ni] = __builtin_shufflevector(lo, hi, 0,1,2,3,4,5,6,7,8,9,10,11,12,13,14,15);
    }
#pragma unroll
    for (int mi = 0; mi < 2; ++mi)
#pragma unroll
      for (int ni = 0; ni < 4; ++ni)
        acc[mi][ni] = __builtin_amdgcn_wmma_f32_16x16x32_f16(
            false, afr[mi], false, bfr[ni], (short)0, acc[mi][ni], false, false);
    __syncthreads();
  }

  // ---- epilogue: C VGPR layout -> global f32 ----
  const int rhi = (lane >> 4) << 3;
#pragma unroll
  for (int mi = 0; mi < 2; ++mi)
#pragma unroll
    for (int ni = 0; ni < 4; ++ni)
#pragma unroll
      for (int r = 0; r < 8; ++r) {
        float v = acc[mi][ni][r] * p.alpha;
        if (EPI == 1)      v = fmaxf(v, 0.f);
        else if (EPI == 2) v = v / (1.f + __expf(-v));
        int row = m0 + wm * 32 + mi * 16 + rhi + r;
        int col = n0 + wn * 64 + ni * 16 + lrow;
        C[(long long)row * p.ldc + col] = v;
      }
}

// ---------------------------------------------------------------------------
// Bottleneck (einsum over 16 channels) + partial BN stats (deterministic)
// y layout: (b, bin, w) with w fastest.
// ---------------------------------------------------------------------------
__global__ __launch_bounds__(256) void bottleneck_kernel(const float* __restrict__ x,
                                                         const float* __restrict__ wv,
                                                         float* __restrict__ y,
                                                         float* __restrict__ part) {
  __shared__ float rs[256], rq[256];
  int idx = blockIdx.x * 256 + threadIdx.x;   // < 4*1024*512
  int w   = idx & 511;
  int bin = (idx >> 9) & 1023;
  int b   = idx >> 19;
  float acc = 0.f;
#pragma unroll
  for (int c = 0; c < 16; ++c)
    acc += x[(((long long)(b * 16 + c)) * 1024 + bin) * 512 + w] * wv[c];
  y[idx] = acc;
  rs[threadIdx.x] = acc;
  rq[threadIdx.x] = acc * acc;
  __syncthreads();
  for (int off = 128; off > 0; off >>= 1) {
    if (threadIdx.x < off) {
      rs[threadIdx.x] += rs[threadIdx.x + off];
      rq[threadIdx.x] += rq[threadIdx.x + off];
    }
    __syncthreads();
  }
  if (threadIdx.x == 0) { part[2 * blockIdx.x] = rs[0]; part[2 * blockIdx.x + 1] = rq[0]; }
}

__global__ __launch_bounds__(256) void reduce_stats_kernel(const float* __restrict__ part,
                                                           int nblk, float* __restrict__ stats) {
  __shared__ float rs[256], rq[256];
  float s = 0.f, q = 0.f;
  for (int i = threadIdx.x; i < nblk; i += 256) { s += part[2 * i]; q += part[2 * i + 1]; }
  rs[threadIdx.x] = s; rq[threadIdx.x] = q;
  __syncthreads();
  for (int off = 128; off > 0; off >>= 1) {
    if (threadIdx.x < off) {
      rs[threadIdx.x] += rs[threadIdx.x + off];
      rq[threadIdx.x] += rq[threadIdx.x + off];
    }
    __syncthreads();
  }
  if (threadIdx.x == 0) { stats[0] = rs[0]; stats[1] = rq[0]; }
}

// BN normalize + ReLU + transpose (b,bin,w) -> (b,w,bin)
__global__ __launch_bounds__(256) void bn_transpose_kernel(const float* __restrict__ y,
                                                           const float* __restrict__ stats,
                                                           const float* __restrict__ g,
                                                           const float* __restrict__ bt,
                                                           float* __restrict__ xact) {
  int idx = blockIdx.x * 256 + threadIdx.x;   // (b, w, bin), bin fastest
  int bin = idx & 1023;
  int w   = (idx >> 10) & 511;
  int b   = idx >> 19;
  const float cnt = 4.f * 1024.f * 512.f;
  float m   = stats[0] / cnt;
  float var = stats[1] / cnt - m * m;
  float v = (y[((long long)(b * 1024 + bin)) * 512 + w] - m) * rsqrtf(var + 1e-5f) * g[0] + bt[0];
  xact[idx] = fmaxf(v, 0.f);
}

// ---------------------------------------------------------------------------
// Row LayerNorm: out = LN(base + add1 + zero_pad(add2)) * g + beta
// One block per row (row = b*512 + w), N in {1024, 4096}.
// ---------------------------------------------------------------------------
__global__ __launch_bounds__(256) void ln_kernel(float* __restrict__ out,
                                                 const float* __restrict__ base,
                                                 const float* __restrict__ add1,
                                                 const float* __restrict__ add2, int n2,
                                                 const float* __restrict__ g,
                                                 const float* __restrict__ beta, int N) {
  __shared__ float vals[4096];
  __shared__ float red[256];
  long long row = blockIdx.x;
  const float* bp = base + row * N;
  const float* a1 = add1 ? add1 + row * N  : nullptr;
  const float* a2 = add2 ? add2 + row * n2 : nullptr;
  float s = 0.f;
  for (int j = threadIdx.x; j < N; j += 256) {
    float v = bp[j];
    if (a1) v += a1[j];
    if (a2 && j < n2) v += a2[j];
    vals[j] = v; s += v;
  }
  red[threadIdx.x] = s; __syncthreads();
  for (int off = 128; off > 0; off >>= 1) {
    if (threadIdx.x < off) red[threadIdx.x] += red[threadIdx.x + off];
    __syncthreads();
  }
  float mean = red[0] / N;
  __syncthreads();
  float q = 0.f;
  for (int j = threadIdx.x; j < N; j += 256) { float d = vals[j] - mean; q += d * d; }
  red[threadIdx.x] = q; __syncthreads();
  for (int off = 128; off > 0; off >>= 1) {
    if (threadIdx.x < off) red[threadIdx.x] += red[threadIdx.x + off];
    __syncthreads();
  }
  float inv = rsqrtf(red[0] / N + 1e-5f);
  float* op = out + row * N;
  for (int j = threadIdx.x; j < N; j += 256)
    op[j] = (vals[j] - mean) * inv * g[j] + beta[j];
}

// ---------------------------------------------------------------------------
// Attention logits + softmax. skew reduces to qe^T (since w == cs == 512):
//   logits[z,i,j] = qk[z,i,j] (pre-scaled) + 0.5*|i-j|*dw[n,i] + qe[z,j,i]
// ---------------------------------------------------------------------------
__global__ __launch_bounds__(256) void attn_softmax_kernel(float* __restrict__ L,
                                                           const float* __restrict__ qe,
                                                           const float* __restrict__ dw) {
  const int W = 512;
  __shared__ float vals[512];
  __shared__ float red[256];
  int bi = blockIdx.x;
  int i  = bi & (W - 1);
  int z  = bi >> 9;
  int n  = z & 3;
  float* row = L + ((long long)z * W + i) * W;
  const float* qz = qe + (long long)z * W * W + i;   // + j*W
  float dwv = dw[n * W + i];
  float mx = -1e30f;
  for (int j = threadIdx.x; j < W; j += 256) {
    float v = row[j] + 0.5f * fabsf((float)(i - j)) * dwv + qz[(long long)j * W];
    vals[j] = v;
    mx = fmaxf(mx, v);
  }
  red[threadIdx.x] = mx; __syncthreads();
  for (int off = 128; off > 0; off >>= 1) {
    if (threadIdx.x < off) red[threadIdx.x] = fmaxf(red[threadIdx.x], red[threadIdx.x + off]);
    __syncthreads();
  }
  mx = red[0]; __syncthreads();
  float s = 0.f;
  for (int j = threadIdx.x; j < W; j += 256) {
    float e = __expf(vals[j] - mx);
    vals[j] = e; s += e;
  }
  red[threadIdx.x] = s; __syncthreads();
  for (int off = 128; off > 0; off >>= 1) {
    if (threadIdx.x < off) red[threadIdx.x] += red[threadIdx.x + off];
    __syncthreads();
  }
  float inv = 1.f / red[0];
  for (int j = threadIdx.x; j < W; j += 256) row[j] = vals[j] * inv;
}

// ---------------------------------------------------------------------------
// Depthwise conv1d along w on (b, w, C) layout (channel-coalesced).
// ---------------------------------------------------------------------------
__global__ __launch_bounds__(256) void dwconv_kernel(const float* __restrict__ in,
                                                     const float* __restrict__ f,
                                                     float* __restrict__ out,
                                                     int W, int C, int K, int P) {
  long long idx = (long long)blockIdx.x * 256 + threadIdx.x;
  int c = (int)(idx % C);
  long long t = idx / C;
  int w = (int)(t % W);
  int b = (int)(t / W);
  float acc = 0.f;
  for (int k = 0; k < K; ++k) {
    int ww = w + k - P;
    if (ww >= 0 && ww < W)
      acc += in[((long long)(b * W + ww)) * C + c] * f[c * K + k];
  }
  out[idx] = acc;
}

// Final (b,w,bins) -> (b,1,bins,w)
__global__ __launch_bounds__(256) void out_transpose_kernel(const float* __restrict__ x,
                                                            float* __restrict__ out) {
  int idx = blockIdx.x * 256 + threadIdx.x;   // (b, bin, w), w fastest
  int w   = idx & 511;
  int bin = (idx >> 9) & 1023;
  int b   = idx >> 19;
  out[idx] = x[((long long)b * 512 + w) * 1024 + bin];
}

// ---------------------------------------------------------------------------
// Host orchestration
// ---------------------------------------------------------------------------
extern "C" void kernel_launch(void* const* d_in, const int* in_sizes, int n_in,
                              void* d_out, int out_size, void* d_ws, size_t ws_size,
                              hipStream_t stream) {
  (void)in_sizes; (void)n_in; (void)out_size; (void)ws_size;
  const float* in_x     = (const float*)d_in[0];
  const float* in_mem   = (const float*)d_in[1];
  const float* bott_w   = (const float*)d_in[2];
  const float* bn1_g    = (const float*)d_in[3];
  const float* bn1_b    = (const float*)d_in[4];
  const float* mbott_w  = (const float*)d_in[5];
  const float* bn2_g    = (const float*)d_in[6];
  const float* bn2_b    = (const float*)d_in[7];
  const float* sa1p[6]; const float* ea1p[6]; const float* sa2p[6]; const float* ea2p[6];
  for (int i = 0; i < 6; ++i) sa1p[i] = (const float*)d_in[8 + i];
  for (int i = 0; i < 6; ++i) ea1p[i] = (const float*)d_in[14 + i];
  for (int i = 0; i < 6; ++i) sa2p[i] = (const float*)d_in[20 + i];
  for (int i = 0; i < 6; ++i) ea2p[i] = (const float*)d_in[26 + i];
  const float* ln1_g = (const float*)d_in[32]; const float* ln1_b = (const float*)d_in[33];
  const float* ln2_g = (const float*)d_in[34]; const float* ln2_b = (const float*)d_in[35];
  const float* ln3_g = (const float*)d_in[36]; const float* ln3_b = (const float*)d_in[37];
  const float* ln4_g = (const float*)d_in[38]; const float* ln4_b = (const float*)d_in[39];
  const float* ln5_g = (const float*)d_in[40]; const float* ln5_b = (const float*)d_in[41];
  const float* ln6_g = (const float*)d_in[42]; const float* ln6_b = (const float*)d_in[43];
  const float* c1L_dw = (const float*)d_in[44];
  const float* c1L_pw = (const float*)d_in[45];
  const float* c1R_dw = (const float*)d_in[46];
  const float* c1R_pw = (const float*)d_in[47];
  const float* c2_dw  = (const float*)d_in[48];
  const float* c2_pw  = (const float*)d_in[49];
  const float* c3_w   = (const float*)d_in[50];
  const float* c4_w   = (const float*)d_in[51];

  // Workspace layout (floats)
  const size_t P2M = 2097152ULL;      // 4*512*1024
  const size_t P4M = 4194304ULL;      // 4*4*512*512
  const size_t P8M = 8388608ULL;      // 4*512*4096
  float* W      = (float*)d_ws;
  float* xact   = W;                  // running x (b, w, 1024)
  float* memact = xact + P2M;
  float* qb     = memact + P2M;       // q projections / o_all / bottleneck scratch
  float* kb     = qb + P2M;
  float* vb     = kb + P2M;
  float* t1     = vb + P2M;
  float* t2     = t1 + P2M;
  float* qe     = t2 + P2M;           // (b,nb,512,512)
  float* Lb     = qe + P4M;           // logits -> probs
  float* big1   = Lb + P4M;           // (b,w,4096)
  float* big2   = big1 + P8M;
  float* stats  = big2 + P8M;         // 4 floats
  float* part   = stats + 8;          // 2*8192 partials

  auto gemm = [&](const float* A, const float* Bm, float* C,
                  int M, int N, int K, int lda, int ldb, int ldc,
                  int Z, int zdiv,
                  long long sA1, long long sA2, long long sB1, long long sB2,
                  long long sC1, long long sC2, int bT, int epi, float alpha) {
    GemmP p;
    p.A = A; p.B = Bm; p.C = C;
    p.M = M; p.N = N; p.K = K;
    p.lda = lda; p.ldb = ldb; p.ldc = ldc;
    p.zdiv = zdiv;
    p.sA1 = sA1; p.sA2 = sA2; p.sB1 = sB1; p.sB2 = sB2; p.sC1 = sC1; p.sC2 = sC2;
    p.bTrans = bT; p.epi = epi; p.alpha = alpha;
    dim3 grid(N / TBN, M / TBM, Z);
    if (bT) {
      if (epi == 0)      gemm_wmma_kernel<1, 0><<<grid, 256, 0, stream>>>(p);
      else if (epi == 1) gemm_wmma_kernel<1, 1><<<grid, 256, 0, stream>>>(p);
      else               gemm_wmma_kernel<1, 2><<<grid, 256, 0, stream>>>(p);
    } else {
      gemm_wmma_kernel<0, 0><<<grid, 256, 0, stream>>>(p);
    }
  };

  auto run_attn = [&](const float* xin, const float* min_, const float* const* ap, float* outb) {
    const float* qw = ap[0]; const float* kw = ap[1]; const float* vw = ap[2];
    const float* ow = ap[3]; const float* er = ap[4]; const float* dw = ap[5];
    gemm(xin,  qw, qb, 2048, 1024, 1024, 1024, 1024, 1024, 1, 1, 0,0, 0,0, 0,0, 1, 0, 1.f);
    gemm(min_, kw, kb, 2048, 1024, 1024, 1024, 1024, 1024, 1, 1, 0,0, 0,0, 0,0, 1, 0, 1.f);
    gemm(min_, vw, vb, 2048, 1024, 1024, 1024, 1024, 1024, 1, 1, 0,0, 0,0, 0,0, 1, 0, 1.f);
    // qk[z] = q_head @ k_head^T / 32   (z = b*4+n, zdiv=4)
    gemm(qb, kb, Lb, 512, 512, 256, 1024, 1024, 512, 16, 4,
         524288, 256, 524288, 256, 1048576, 262144, 1, 0, 0.03125f);
    // qe[z] = q_head @ er   (er is (d=256, c=512) K-major)
    gemm(qb, er, qe, 512, 512, 256, 1024, 512, 512, 16, 4,
         524288, 256, 0, 0, 1048576, 262144, 0, 0, 1.f);
    attn_softmax_kernel<<<16 * 512, 256, 0, stream>>>(Lb, qe, dw);
    // o_head = probs @ v_head  -> qb reused as o_all
    gemm(Lb, vb, qb, 512, 256, 512, 512, 1024, 1024, 16, 4,
         1048576, 262144, 524288, 256, 524288, 256, 0, 0, 1.f);
    gemm(qb, ow, outb, 2048, 1024, 1024, 1024, 1024, 1024, 1, 1, 0,0, 0,0, 0,0, 1, 0, 1.f);
  };

  // ---- 1) bottlenecks + BN + ReLU + transpose ----
  bottleneck_kernel<<<8192, 256, 0, stream>>>(in_x, bott_w, qb, part);
  reduce_stats_kernel<<<1, 256, 0, stream>>>(part, 8192, stats);
  bn_transpose_kernel<<<8192, 256, 0, stream>>>(qb, stats, bn1_g, bn1_b, xact);
  bottleneck_kernel<<<8192, 256, 0, stream>>>(in_mem, mbott_w, qb, part);
  reduce_stats_kernel<<<1, 256, 0, stream>>>(part, 8192, stats + 2);
  bn_transpose_kernel<<<8192, 256, 0, stream>>>(qb, stats + 2, bn2_g, bn2_b, memact);

  // ---- 2) sa1 + ea1 + ln1 ----
  run_attn(xact, xact,   sa1p, t1);
  run_attn(xact, memact, ea1p, t2);
  ln_kernel<<<2048, 256, 0, stream>>>(xact, xact, t1, t2, 1024, ln1_g, ln1_b, 1024);

  // ---- 3) conv block ----
  dwconv_kernel<<<8192, 256, 0, stream>>>(xact, c1L_dw, t1, 512, 1024, 11, 5);
  gemm(t1, c1L_pw, big1, 2048, 4096, 1024, 1024, 1024, 4096, 1, 1, 0,0, 0,0, 0,0, 1, 1, 1.f); // relu
  dwconv_kernel<<<8192, 256, 0, stream>>>(xact, c1R_dw, t2, 512, 1024, 7, 3);
  gemm(t2, c1R_pw, t1, 2048, 1024, 1024, 1024, 1024, 1024, 1, 1, 0,0, 0,0, 0,0, 1, 0, 1.f);
  ln_kernel<<<2048, 256, 0, stream>>>(big1, big1, nullptr, t1, 1024, ln2_g, ln2_b, 4096);
  dwconv_kernel<<<32768, 256, 0, stream>>>(big1, c2_dw, big2, 512, 4096, 7, 3);
  gemm(big2, c2_pw, t1, 2048, 1024, 4096, 4096, 4096, 1024, 1, 1, 0,0, 0,0, 0,0, 1, 0, 1.f);
  ln_kernel<<<2048, 256, 0, stream>>>(xact, xact, t1, nullptr, 0, ln3_g, ln3_b, 1024);

  // ---- 4) sa2 + ln4 ----
  run_attn(xact, xact, sa2p, t1);
  ln_kernel<<<2048, 256, 0, stream>>>(xact, xact, t1, nullptr, 0, ln4_g, ln4_b, 1024);

  // ---- 5) ea2 + ln5 ----
  run_attn(xact, memact, ea2p, t1);
  ln_kernel<<<2048, 256, 0, stream>>>(xact, xact, t1, nullptr, 0, ln5_g, ln5_b, 1024);

  // ---- 6) SiLU MLP + ln6 ----
  gemm(xact, c3_w, big1, 2048, 4096, 1024, 1024, 1024, 4096, 1, 1, 0,0, 0,0, 0,0, 1, 2, 1.f); // silu
  gemm(big1, c4_w, t1, 2048, 1024, 4096, 4096, 4096, 1024, 1, 1, 0,0, 0,0, 0,0, 1, 0, 1.f);
  ln_kernel<<<2048, 256, 0, stream>>>(xact, xact, t1, nullptr, 0, ln6_g, ln6_b, 1024);

  // ---- 7) output transpose ----
  out_transpose_kernel<<<8192, 256, 0, stream>>>(xact, (float*)d_out);
}
